// TransformerDecoderLayer_71476845740838
// MI455X (gfx1250) — matrix operand order repaired and checked by reading, not compiled, not run on hardware
//
#include <hip/hip_runtime.h>

// ---------------------------------------------------------------------------
// Types for CDNA5 WMMA (wave32, 16x16x32 bf16 -> f32)
// ---------------------------------------------------------------------------
typedef __attribute__((ext_vector_type(16))) __bf16 v16bf;
typedef __attribute__((ext_vector_type(8)))  __bf16 v8bf;
typedef __attribute__((ext_vector_type(8)))  float  v8f;

union FragAB { v16bf v; v8bf h[2]; };

__device__ __forceinline__ __bf16 tobf16(float f) { return (__bf16)f; }

// CDNA5 async memory->LDS copy (16B per lane), tracked by ASYNCcnt.
// vdst = per-lane LDS byte offset, vaddr = per-lane 64-bit global address.
__device__ __forceinline__ void async_copy_b128(unsigned lds_off,
                                                unsigned long long gaddr) {
  asm volatile("global_load_async_to_lds_b128 %0, %1, off"
               :: "v"(lds_off), "v"(gaddr) : "memory");
}
__device__ __forceinline__ void wait_async_le(int keep4) {
  if (keep4) asm volatile("s_wait_asynccnt 0x4" ::: "memory");
  else       asm volatile("s_wait_asynccnt 0x0" ::: "memory");
}
__device__ __forceinline__ unsigned lds_off_u32(const void* p) {
  return (unsigned)(unsigned long long)p;  // low 32 bits of generic = LDS offset
}

// ---------------------------------------------------------------------------
// fp32 -> bf16 cast (grid-stride)
// ---------------------------------------------------------------------------
__global__ void cast_f32_bf16(const float* __restrict__ src,
                              __bf16* __restrict__ dst, long long n) {
  long long i = (long long)blockIdx.x * blockDim.x + threadIdx.x;
  long long stride = (long long)gridDim.x * blockDim.x;
  for (; i < n; i += stride) dst[i] = tobf16(src[i]);
}

// ---------------------------------------------------------------------------
// Tiled bf16 GEMM with WMMA:  C[M,N] = A[M,K] * B[N,K]^T  (+ epilogue)
//   Block tile 128x128, K-step 32, 256 threads = 8 waves, wave tile 32x64.
//   Double-buffered LDS fed by async global->LDS copies (no VGPR staging).
//   Epilogues (EPI):
//     0: bf16 store (+bias)                [row-major, batch stride sC]
//     1: bf16 transposed store (+bias)     -> out[b*N*S + n*S + s], b=m/S
//     2: f32 store (raw, no bias)          [scores -> d_out]
//     3: f32 store (+bias +residual)       [pre-LayerNorm]
//     4: relu -> bf16 (+bias)              [FF hidden]
// ---------------------------------------------------------------------------
template <int EPI>
__global__ __launch_bounds__(256)
void gemm_bf16_wmma(const __bf16* __restrict__ A, const __bf16* __restrict__ Bw,
                    const float* __restrict__ bias, void* __restrict__ Cout,
                    const float* __restrict__ Res,
                    int N, int K, int lda, int ldb, int ldc,
                    long long sA, long long sB, long long sC, long long sRes,
                    int Sseq) {
  constexpr int APITCH = 40;  // 32 + 8 bf16 pad (80B rows, 16B aligned)
  __shared__ __bf16 As[2][128 * APITCH];
  __shared__ __bf16 Bs[2][128 * APITCH];

  A  += (long long)blockIdx.z * sA;
  Bw += (long long)blockIdx.z * sB;
  const float* res = Res ? (Res + (long long)blockIdx.z * sRes) : nullptr;

  const int tid  = threadIdx.x;
  const int lane = tid & 31;
  const int wave = tid >> 5;
  const int wm   = wave & 3;   // wave row: 4 x 32 rows
  const int wn   = wave >> 2;  // wave col: 2 x 64 cols
  const int rowBase = blockIdx.y * 128;
  const int colBase = blockIdx.x * 128;

  // cooperative tile-load indexing: 256 thr x 16 bf16 = 128x32 tile
  const int ldRow = tid >> 1;
  const int ldCol = (tid & 1) << 4;

  // per-thread global source addresses (advance by 32 elems per K-step)
  const __bf16* ga = A  + (long long)(rowBase + ldRow) * lda + ldCol;
  const __bf16* gb = Bw + (long long)(colBase + ldRow) * ldb + ldCol;
  // per-thread LDS destinations for the two buffers
  const unsigned ldsStage = (unsigned)(ldRow * APITCH + ldCol) * 2u;
  unsigned ldsA[2], ldsB[2];
  ldsA[0] = lds_off_u32(&As[0][0]) + ldsStage;
  ldsA[1] = lds_off_u32(&As[1][0]) + ldsStage;
  ldsB[0] = lds_off_u32(&Bs[0][0]) + ldsStage;
  ldsB[1] = lds_off_u32(&Bs[1][0]) + ldsStage;

  auto stage = [&](int buf, int k0) {
    const __bf16* pa = ga + k0;
    const __bf16* pb = gb + k0;
    async_copy_b128(ldsA[buf],      (unsigned long long)pa);
    async_copy_b128(ldsA[buf] + 16, (unsigned long long)(pa + 8));
    async_copy_b128(ldsB[buf],      (unsigned long long)pb);
    async_copy_b128(ldsB[buf] + 16, (unsigned long long)(pb + 8));
  };

  v8f zero;
  #pragma unroll
  for (int e = 0; e < 8; ++e) zero[e] = 0.0f;
  v8f acc[2][4];
  #pragma unroll
  for (int i = 0; i < 2; ++i)
    #pragma unroll
    for (int j = 0; j < 4; ++j) acc[i][j] = zero;

  const int m16    = lane & 15;
  const int half   = lane >> 4;      // 0 or 1
  const int khalfA = half * 8;       // A: element K base per lane group
  const int koffB  = half * 8;       // B: element K base per lane group

  stage(0, 0);
  for (int k0 = 0; k0 < K; k0 += 32) {
    const int cur = (k0 >> 5) & 1;
    const bool more = (k0 + 32) < K;
    if (more) stage(cur ^ 1, k0 + 32);   // prefetch next slice into other buffer
    wait_async_le(more ? 1 : 0);         // current slice resident in LDS
    __syncthreads();

    // A fragments: 16x32 tiles, lane m = lane&15, K split by lane half
    FragAB a[2];
    #pragma unroll
    for (int i = 0; i < 2; ++i) {
      const __bf16* pa = &As[cur][(wm * 32 + i * 16 + m16) * APITCH + khalfA];
      a[i].h[0] = *(const v8bf*)pa;         // e0..7  : K = khalf*8 + 0..7
      a[i].h[1] = *(const v8bf*)(pa + 16);  // e8..15 : K = 16 + khalf*8 + 0..7
    }
    // B fragments: 32x16 tiles, lane n = lane&15, K split by lane half (same as A)
    FragAB b[4];
    #pragma unroll
    for (int j = 0; j < 4; ++j) {
      const __bf16* pb = &Bs[cur][(wn * 64 + j * 16 + m16) * APITCH + koffB];
      b[j].h[0] = *(const v8bf*)pb;
      b[j].h[1] = *(const v8bf*)(pb + 16);
    }
    #pragma unroll
    for (int i = 0; i < 2; ++i)
      #pragma unroll
      for (int j = 0; j < 4; ++j)
        acc[i][j] = __builtin_amdgcn_wmma_f32_16x16x32_bf16(
            false, a[i].v, false, b[j].v, (short)0, acc[i][j], false, false);
    __syncthreads();  // all reads of 'cur' done before it is re-staged
  }

  // epilogue; C/D layout: elem e -> M = e + 8*(lane>>4), N = lane&15
  #pragma unroll
  for (int i = 0; i < 2; ++i) {
    #pragma unroll
    for (int j = 0; j < 4; ++j) {
      #pragma unroll
      for (int e = 0; e < 8; ++e) {
        const int r = rowBase + wm * 32 + i * 16 + half * 8 + e;
        const int c = colBase + wn * 64 + j * 16 + m16;
        float val = acc[i][j][e];
        if constexpr (EPI == 0) {
          if (bias) val += bias[c];
          __bf16* Cb = (__bf16*)Cout + (long long)blockIdx.z * sC;
          Cb[(long long)r * ldc + c] = tobf16(val);
        } else if constexpr (EPI == 1) {
          if (bias) val += bias[c];
          const long long bI = r / Sseq, s = r % Sseq;
          ((__bf16*)Cout)[(bI * N + c) * Sseq + s] = tobf16(val);
        } else if constexpr (EPI == 2) {
          float* Cf = (float*)Cout + (long long)blockIdx.z * sC;
          Cf[(long long)r * ldc + c] = val;
        } else if constexpr (EPI == 3) {
          val += bias[c] + res[(long long)r * ldc + c];
          float* Cf = (float*)Cout + (long long)blockIdx.z * sC;
          Cf[(long long)r * ldc + c] = val;
        } else {  // EPI == 4
          val += bias[c];
          val = fmaxf(val, 0.0f);
          __bf16* Cb = (__bf16*)Cout + (long long)blockIdx.z * sC;
          Cb[(long long)r * ldc + c] = tobf16(val);
        }
      }
    }
  }
}

// ---------------------------------------------------------------------------
// Row softmax with scale + optional causal mask.
// In-place on fp32 scores (d_out region), bf16 copy for the context GEMM.
// One 256-thread block per row; S = 2048 -> 8 columns per thread.
// ---------------------------------------------------------------------------
__global__ __launch_bounds__(256)
void softmax_rows(float* __restrict__ sc, __bf16* __restrict__ ob,
                  int S, int causal, int qmod, float scale) {
  __shared__ float red[256];
  const int tid = threadIdx.x;
  const long long row = blockIdx.x;
  const int q = (int)(row % qmod);
  float* p = sc + row * (long long)S;

  float v[8];
  float mx = -3.4e38f;
  #pragma unroll
  for (int k = 0; k < 8; ++k) {
    const int c = tid + k * 256;
    float t = p[c] * scale;
    if (causal && c > q) t = -3.4e38f;
    v[k] = t;
    mx = fmaxf(mx, t);
  }
  red[tid] = mx; __syncthreads();
  for (int s = 128; s > 0; s >>= 1) {
    if (tid < s) red[tid] = fmaxf(red[tid], red[tid + s]);
    __syncthreads();
  }
  mx = red[0]; __syncthreads();

  float sum = 0.0f;
  #pragma unroll
  for (int k = 0; k < 8; ++k) {
    float e = (v[k] <= -3.0e38f) ? 0.0f : __expf(v[k] - mx);
    v[k] = e; sum += e;
  }
  red[tid] = sum; __syncthreads();
  for (int s = 128; s > 0; s >>= 1) {
    if (tid < s) red[tid] += red[tid + s];
    __syncthreads();
  }
  const float inv = 1.0f / red[0];

  #pragma unroll
  for (int k = 0; k < 8; ++k) {
    const int c = tid + k * 256;
    const float pr = v[k] * inv;
    p[c] = pr;
    ob[row * (long long)S + c] = tobf16(pr);
  }
}

// ---------------------------------------------------------------------------
// Row LayerNorm over D=1024: fp32 out (+ optional bf16 copy).
// ---------------------------------------------------------------------------
__global__ __launch_bounds__(256)
void layernorm_rows(const float* __restrict__ in, const float* __restrict__ g,
                    const float* __restrict__ b, float* __restrict__ outf,
                    __bf16* __restrict__ outb, int D) {
  __shared__ float red[256];
  const int tid = threadIdx.x;
  const long long row = blockIdx.x;
  const float* p = in + row * (long long)D;

  float v[4];
  float s = 0.0f;
  #pragma unroll
  for (int k = 0; k < 4; ++k) { v[k] = p[tid + k * 256]; s += v[k]; }
  red[tid] = s; __syncthreads();
  for (int t = 128; t > 0; t >>= 1) {
    if (tid < t) red[tid] += red[tid + t];
    __syncthreads();
  }
  const float mu = red[0] / (float)D; __syncthreads();

  float ss = 0.0f;
  #pragma unroll
  for (int k = 0; k < 4; ++k) { const float d = v[k] - mu; ss += d * d; }
  red[tid] = ss; __syncthreads();
  for (int t = 128; t > 0; t >>= 1) {
    if (tid < t) red[tid] += red[tid + t];
    __syncthreads();
  }
  const float rstd = rsqrtf(red[0] / (float)D + 1e-5f);

  #pragma unroll
  for (int k = 0; k < 4; ++k) {
    const int c = tid + k * 256;
    const float y = (v[k] - mu) * rstd * g[c] + b[c];
    outf[row * (long long)D + c] = y;
    if (outb) outb[row * (long long)D + c] = tobf16(y);
  }
}

// ---------------------------------------------------------------------------
// Host-side orchestration
// ---------------------------------------------------------------------------
extern "C" void kernel_launch(void* const* d_in, const int* in_sizes, int n_in,
                              void* d_out, int out_size, void* d_ws, size_t ws_size,
                              hipStream_t stream) {
  constexpr int B = 4, ST = 2048, SS = 2048, D = 1024, DFF = 4096;
  constexpr long long M1 = (long long)B * ST;  // 8192 query rows
  (void)in_sizes; (void)n_in; (void)out_size; (void)ws_size;

  const float* x     = (const float*)d_in[0];
  const float* enc   = (const float*)d_in[1];
  // d_in[2] = tgt_mask (tril) -> applied analytically as causal mask
  const float* sa_wq = (const float*)d_in[3];  const float* sa_bq = (const float*)d_in[4];
  const float* sa_wk = (const float*)d_in[5];  const float* sa_bk = (const float*)d_in[6];
  const float* sa_wv = (const float*)d_in[7];  const float* sa_bv = (const float*)d_in[8];
  const float* sa_wo = (const float*)d_in[9];  const float* sa_bo = (const float*)d_in[10];
  const float* ca_wq = (const float*)d_in[11]; const float* ca_bq = (const float*)d_in[12];
  const float* ca_wk = (const float*)d_in[13]; const float* ca_bk = (const float*)d_in[14];
  const float* ca_wv = (const float*)d_in[15]; const float* ca_bv = (const float*)d_in[16];
  const float* ca_wo = (const float*)d_in[17]; const float* ca_bo = (const float*)d_in[18];
  const float* ln1_g = (const float*)d_in[19]; const float* ln1_b = (const float*)d_in[20];
  const float* ln2_g = (const float*)d_in[21]; const float* ln2_b = (const float*)d_in[22];
  const float* ln3_g = (const float*)d_in[23]; const float* ln3_b = (const float*)d_in[24];
  const float* ff_w1 = (const float*)d_in[25]; const float* ff_b1 = (const float*)d_in[26];
  const float* ff_w2 = (const float*)d_in[27]; const float* ff_b2 = (const float*)d_in[28];

  float* out_x          = (float*)d_out;                           // (B,ST,D)
  float* out_attn_self  = out_x + M1 * D;                          // (B,ST,ST)
  float* out_attn_cross = out_attn_self + (long long)B * ST * ST;  // (B,ST,SS)

  // ---- carve workspace (256B aligned) ----
  char* base = (char*)d_ws;
  size_t off = 0;
  auto alloc = [&](size_t bytes) -> void* {
    void* p = base + off;
    off += (bytes + 255) & ~(size_t)255;
    return p;
  };
  const size_t D2 = (size_t)D * D;
  __bf16* wqb  = (__bf16*)alloc(D2 * 2);
  __bf16* wkb  = (__bf16*)alloc(D2 * 2);
  __bf16* wvb  = (__bf16*)alloc(D2 * 2);
  __bf16* wob  = (__bf16*)alloc(D2 * 2);
  __bf16* cwqb = (__bf16*)alloc(D2 * 2);
  __bf16* cwkb = (__bf16*)alloc(D2 * 2);
  __bf16* cwvb = (__bf16*)alloc(D2 * 2);
  __bf16* cwob = (__bf16*)alloc(D2 * 2);
  __bf16* w1b  = (__bf16*)alloc((size_t)DFF * D * 2);
  __bf16* w2b  = (__bf16*)alloc((size_t)D * DFF * 2);
  __bf16* xb   = (__bf16*)alloc((size_t)M1 * D * 2);
  __bf16* encb = (__bf16*)alloc((size_t)M1 * D * 2);
  __bf16* qb   = (__bf16*)alloc((size_t)M1 * D * 2);
  __bf16* kb   = (__bf16*)alloc((size_t)M1 * D * 2);
  __bf16* vt   = (__bf16*)alloc((size_t)B * D * ST * 2);   // V^T per batch (D x S)
  __bf16* attnb= (__bf16*)alloc((size_t)B * ST * ST * 2);
  __bf16* ctxb = (__bf16*)alloc((size_t)M1 * D * 2);
  __bf16* hb   = (__bf16*)alloc((size_t)M1 * DFF * 2);
  __bf16* x1b  = (__bf16*)alloc((size_t)M1 * D * 2);
  __bf16* x2b  = (__bf16*)alloc((size_t)M1 * D * 2);
  float*  tmpf = (float*)alloc((size_t)M1 * D * 4);
  float*  x1f  = (float*)alloc((size_t)M1 * D * 4);
  float*  x2f  = (float*)alloc((size_t)M1 * D * 4);

  // ---- cast fp32 -> bf16 ----
  auto cast = [&](const float* s, __bf16* d, long long n) {
    int grid = (int)((n + 256 * 8 - 1) / (256 * 8));
    if (grid > 8192) grid = 8192;
    cast_f32_bf16<<<grid, 256, 0, stream>>>(s, d, n);
  };
  cast(x, xb, M1 * D);        cast(enc, encb, M1 * D);
  cast(sa_wq, wqb, D2);       cast(sa_wk, wkb, D2);
  cast(sa_wv, wvb, D2);       cast(sa_wo, wob, D2);
  cast(ca_wq, cwqb, D2);      cast(ca_wk, cwkb, D2);
  cast(ca_wv, cwvb, D2);      cast(ca_wo, cwob, D2);
  cast(ff_w1, w1b, (long long)DFF * D);
  cast(ff_w2, w2b, (long long)D * DFF);

  const dim3 blk(256);
  const float scale = 0.03125f;  // 1/sqrt(1024)
  const long long sQK = (long long)ST * D;
  const long long sSC = (long long)ST * ST;
  const long long sVT = (long long)D * ST;

  // ================= self-attention =================
  gemm_bf16_wmma<0><<<dim3(8, 64, 1), blk, 0, stream>>>(
      xb, wqb, sa_bq, qb, nullptr, D, D, D, D, D, 0, 0, 0, 0, 0);
  gemm_bf16_wmma<0><<<dim3(8, 64, 1), blk, 0, stream>>>(
      xb, wkb, sa_bk, kb, nullptr, D, D, D, D, D, 0, 0, 0, 0, 0);
  gemm_bf16_wmma<1><<<dim3(8, 64, 1), blk, 0, stream>>>(
      xb, wvb, sa_bv, vt, nullptr, D, D, D, D, 0, 0, 0, 0, 0, ST);
  // scores = Q K^T -> d_out attn_self region (raw)
  gemm_bf16_wmma<2><<<dim3(16, 16, B), blk, 0, stream>>>(
      qb, kb, nullptr, out_attn_self, nullptr, ST, D, D, D, ST,
      sQK, sQK, sSC, 0, 0);
  softmax_rows<<<(int)M1, blk, 0, stream>>>(out_attn_self, attnb, ST, 1, ST, scale);
  // ctx = attn V  (B = V^T, N x K = D x ST)
  gemm_bf16_wmma<0><<<dim3(8, 16, B), blk, 0, stream>>>(
      attnb, vt, nullptr, ctxb, nullptr, D, ST, ST, ST, D,
      sSC, sVT, sQK, 0, 0);
  // out proj + bias + residual(x) -> fp32
  gemm_bf16_wmma<3><<<dim3(8, 64, 1), blk, 0, stream>>>(
      ctxb, wob, sa_bo, tmpf, x, D, D, D, D, D, 0, 0, 0, 0, 0);
  layernorm_rows<<<(int)M1, blk, 0, stream>>>(tmpf, ln1_g, ln1_b, x1f, x1b, D);

  // ================= cross-attention =================
  gemm_bf16_wmma<0><<<dim3(8, 64, 1), blk, 0, stream>>>(
      x1b, cwqb, ca_bq, qb, nullptr, D, D, D, D, D, 0, 0, 0, 0, 0);
  gemm_bf16_wmma<0><<<dim3(8, 64, 1), blk, 0, stream>>>(
      encb, cwkb, ca_bk, kb, nullptr, D, D, D, D, D, 0, 0, 0, 0, 0);
  gemm_bf16_wmma<1><<<dim3(8, 64, 1), blk, 0, stream>>>(
      encb, cwvb, ca_bv, vt, nullptr, D, D, D, D, 0, 0, 0, 0, 0, SS);
  gemm_bf16_wmma<2><<<dim3(16, 16, B), blk, 0, stream>>>(
      qb, kb, nullptr, out_attn_cross, nullptr, SS, D, D, D, SS,
      sQK, sQK, sSC, 0, 0);
  softmax_rows<<<(int)M1, blk, 0, stream>>>(out_attn_cross, attnb, SS, 0, ST, scale);
  gemm_bf16_wmma<0><<<dim3(8, 16, B), blk, 0, stream>>>(
      attnb, vt, nullptr, ctxb, nullptr, D, SS, SS, SS, D,
      sSC, sVT, sQK, 0, 0);
  gemm_bf16_wmma<3><<<dim3(8, 64, 1), blk, 0, stream>>>(
      ctxb, cwob, ca_bo, tmpf, x1f, D, D, D, D, D, 0, 0, 0, 0, 0);
  layernorm_rows<<<(int)M1, blk, 0, stream>>>(tmpf, ln2_g, ln2_b, x2f, x2b, D);

  // ================= feed-forward =================
  gemm_bf16_wmma<4><<<dim3(32, 64, 1), blk, 0, stream>>>(
      x2b, w1b, ff_b1, hb, nullptr, DFF, D, D, D, DFF, 0, 0, 0, 0, 0);
  gemm_bf16_wmma<3><<<dim3(8, 64, 1), blk, 0, stream>>>(
      hb, w2b, ff_b2, tmpf, x2f, D, DFF, DFF, DFF, D, 0, 0, 0, 0, 0);
  layernorm_rows<<<(int)M1, blk, 0, stream>>>(tmpf, ln3_g, ln3_b, out_x, nullptr, D);
}
